// mini2d_so3_5875515261438
// MI455X (gfx1250) — matrix operand order repaired and checked by reading, not compiled, not run on hardware
//
#include <hip/hip_runtime.h>
#include <math.h>

// ---------------- static problem dims ----------------
#define NB    64      // batch (nodes)
#define NCH   512     // total channels (onebody width)
#define NCL   128     // channels per degree
#define REPW  2048    // rep components
#define DFOLD 512     // folded dim (REP/4)

typedef __attribute__((ext_vector_type(16))) _Float16 v16h;
typedef __attribute__((ext_vector_type(8)))  _Float16 v8h;
typedef __attribute__((ext_vector_type(8)))  float    v8f;

__device__ __forceinline__ float swishf(float x) { return x / (1.0f + __expf(-x)); }

// ---------------- Stage A: per-(l,channel) L2 norm over m ----------------
__global__ void repnorm_k(const float* __restrict__ x1d, float* __restrict__ norm) {
    int t = blockIdx.x * blockDim.x + threadIdx.x;   // 0 .. 64*512-1
    int n = t >> 9, c = t & 511;
    int l = c >> 7, j = c & 127;
    const int offs[4] = {0, 128, 512, 1152};
    const float* p = x1d + n * REPW + offs[l] + j;
    int nm = 2 * l + 1;
    float s = 0.0f;
    for (int m = 0; m < nm; ++m) { float v = p[m * NCL]; s += v * v; }
    norm[t] = sqrtf(s);
}

// ---------------- Stage B: BatchNorm1d over batch (train stats) ----------------
__global__ void bn_k(const float* __restrict__ norm, _Float16* __restrict__ nhat) {
    int c = blockIdx.x * blockDim.x + threadIdx.x;
    if (c >= NCH) return;
    float mu = 0.0f;
    for (int n = 0; n < NB; ++n) mu += norm[n * NCH + c];
    mu *= (1.0f / NB);
    float var = 0.0f;
    for (int n = 0; n < NB; ++n) { float d = norm[n * NCH + c] - mu; var += d * d; }
    var *= (1.0f / NB);
    float inv = rsqrtf(var + 1e-5f);
    for (int n = 0; n < NB; ++n)
        nhat[n * NCH + c] = (_Float16)((norm[n * NCH + c] - mu) * inv);
}

// ---------------- gauge tensor: g = x1d / (norm[CIND] + 1e-2) ----------------
__global__ void gauge_k(const float* __restrict__ x1d, const float* __restrict__ norm,
                        _Float16* __restrict__ g) {
    int t = blockIdx.x * blockDim.x + threadIdx.x;   // 0 .. 64*2048-1
    int n = t >> 11, r = t & 2047;
    int l, off;
    if      (r < 128)  { l = 0; off = 0;    }
    else if (r < 512)  { l = 1; off = 128;  }
    else if (r < 1152) { l = 2; off = 512;  }
    else               { l = 3; off = 1152; }
    int c = (l << 7) + ((r - off) & 127);
    g[t] = (_Float16)(x1d[t] / (norm[n * NCH + c] + 1e-2f));
}

// ---------------- fp32 -> fp16 weight conversion ----------------
__global__ void cvt_k(const float* __restrict__ s, _Float16* __restrict__ d, int n) {
    int t = blockIdx.x * blockDim.x + threadIdx.x;
    if (t < n) d[t] = (_Float16)s[t];
}

// ---------------- WMMA fragment helpers (16-bit A 16x32, contiguous K order) ----------------
// A: lane = row (lane&15); K = e + 16*hi (contiguous, matches B)
__device__ __forceinline__ v16h load_a_frag(const _Float16* rowk, int hi) {
    const _Float16* p = rowk + hi * 16;
    v8h lo = *(const v8h*)(p);
    v8h hh = *(const v8h*)(p + 8);
    v16h a;
#pragma unroll
    for (int i = 0; i < 8; ++i) { a[i] = lo[i]; a[i + 8] = hh[i]; }
    return a;
}
// B: lane = col (lane&15); 16 contiguous K values starting at kb + 16*hi

// ---------------- onebody MLP GEMM: out = act(A @ W^T + b) ----------------
// A [64,K] f16, W [Nout,K] f16 row-major (so B of the GEMM is contiguous in K).
// One wave per 16x16 tile.  act=1 -> swish -> f16 out ; act=0 -> linear -> f32 out.
__global__ void mlp_gemm_k(const _Float16* __restrict__ A, const _Float16* __restrict__ W,
                           const float* __restrict__ bias, _Float16* __restrict__ outh,
                           float* __restrict__ outf, int K, int Nout, int act) {
    int tiles_n = Nout >> 4;
    int tm = (blockIdx.x / tiles_n) << 4;
    int tn = (blockIdx.x % tiles_n) << 4;
    int lane = threadIdx.x;
    int r = lane & 15, hi = lane >> 4;
    const _Float16* arow = A + (tm + r) * K;
    const _Float16* wrow = W + (tn + r) * K;
    v8f acc = {};
    for (int kb = 0; kb < K; kb += 32) {
        v16h a = load_a_frag(arow + kb, hi);
        v16h b = *(const v16h*)(wrow + kb + hi * 16);
        acc = __builtin_amdgcn_wmma_f32_16x16x32_f16(false, a, false, b, (short)0, acc,
                                                     false, false);
    }
#pragma unroll
    for (int v = 0; v < 8; ++v) {
        int m = tm + v + hi * 8;
        int o = tn + r;
        float x = acc[v] + bias[o];
        if (act) outh[m * Nout + o] = (_Float16)swishf(x);
        else     outf[m * Nout + o] = x;
    }
}

// ---------------- IELin (per-degree channel mix) + fused scalar_mul ----------------
// For degree l: rows = (n,m) pairs (64*(2l+1)), inner = 128, out = 128.
// Epilogue: x1d_new[n, off + m*128 + o] = acc * nmlp[n, l*128 + o]   (written to d_out)
__global__ void ielin_k(const _Float16* __restrict__ g, const _Float16* __restrict__ Wieh,
                        const float* __restrict__ nmlp, float* __restrict__ x1d_out) {
    int t = blockIdx.x;          // 0..511 tiles
    int ct = t & 7;              // col tile (8 per degree-row)
    int rt = t >> 3;             // global row tile 0..63
    int l, r0, off;
    if      (rt < 4)  { l = 0; r0 = rt;      off = 0;    }
    else if (rt < 16) { l = 1; r0 = rt - 4;  off = 128;  }
    else if (rt < 36) { l = 2; r0 = rt - 16; off = 512;  }
    else              { l = 3; r0 = rt - 36; off = 1152; }
    int nm = 2 * l + 1;
    int lane = threadIdx.x;
    int r = lane & 15, hi = lane >> 4;
    int R = r0 * 16 + r;
    int na = R / nm, ma = R % nm;
    const _Float16* arow = g + na * REPW + off + ma * NCL;
    const _Float16* wrow = Wieh + l * NCL * NCL + (ct * 16 + r) * NCL;
    v8f acc = {};
#pragma unroll
    for (int kb = 0; kb < NCL; kb += 32) {
        v16h a = load_a_frag(arow + kb, hi);
        v16h b = *(const v16h*)(wrow + kb + hi * 16);
        acc = __builtin_amdgcn_wmma_f32_16x16x32_f16(false, a, false, b, (short)0, acc,
                                                     false, false);
    }
#pragma unroll
    for (int v = 0; v < 8; ++v) {
        int Ro = r0 * 16 + v + hi * 8;
        int n = Ro / nm, m = Ro % nm;
        int o = ct * 16 + r;
        x1d_out[n * REPW + off + m * NCL + o] = acc[v] * nmlp[n * NCH + l * NCL + o];
    }
}

// ---------------- bilinear pass 1: a2[n,c] = sum_p f[n,p,c] * sum_q x2d[n,p,q,c]*f[n,q,c]
// One block per (n,p): streams the contiguous 8KB row x2d[n,p,:,:] fully coalesced.
__global__ void bilin1_k(const float4* __restrict__ x2d, const float4* __restrict__ f,
                         float* __restrict__ a2) {
    int b = blockIdx.x;                   // n*512 + p
    int n = b >> 9, p = b & 511;
    int tid = threadIdx.x;
    const float4* row = x2d + (size_t)b * DFOLD;
    const float4* fn  = f + (size_t)n * DFOLD;
    float ax = 0, ay = 0, az = 0, aw = 0;
    for (int q = tid; q < DFOLD; q += 256) {
        float4 x = row[q]; float4 fq = fn[q];
        ax += x.x * fq.x; ay += x.y * fq.y; az += x.z * fq.z; aw += x.w * fq.w;
    }
    __shared__ float4 red[256];
    red[tid] = make_float4(ax, ay, az, aw);
    __syncthreads();
    for (int s = 128; s > 0; s >>= 1) {
        if (tid < s) {
            red[tid].x += red[tid + s].x; red[tid].y += red[tid + s].y;
            red[tid].z += red[tid + s].z; red[tid].w += red[tid + s].w;
        }
        __syncthreads();
    }
    if (tid == 0) {
        float4 fp = fn[p];
        atomicAdd(&a2[n * 4 + 0], red[0].x * fp.x);
        atomicAdd(&a2[n * 4 + 1], red[0].y * fp.y);
        atomicAdd(&a2[n * 4 + 2], red[0].z * fp.z);
        atomicAdd(&a2[n * 4 + 3], red[0].w * fp.w);
    }
}

// ---------------- gate: swish(a2 @ Wg^T + bg) ----------------
__global__ void gate_k(const float* __restrict__ a2, const float* __restrict__ Wg,
                       const float* __restrict__ bg, float* __restrict__ gate) {
    int t = threadIdx.x;
    if (t >= NB * 4) return;
    int n = t >> 2, d = t & 3;
    float s = bg[d];
    for (int c = 0; c < 4; ++c) s += a2[n * 4 + c] * Wg[d * 4 + c];
    gate[t] = swishf(s);
}

// ---------------- bilinear pass 2: x2d_new = (x2d . Wt^T) * gate, streamed ----------------
__global__ void bilin2_k(const float4* __restrict__ x2d, const float* __restrict__ Wt,
                         const float4* __restrict__ gate, float4* __restrict__ out) {
    __shared__ float w[16];
    int tid = threadIdx.x;
    if (tid < 16) w[tid] = Wt[tid];
    __syncthreads();
    size_t idx = (size_t)blockIdx.x * 256 + tid;      // float4 index; n = idx / 2^18
    int n = (int)(idx >> 18);
    float4 x = x2d[idx];
    float4 gv = gate[n];
    float4 y;
    y.x = (w[0]  * x.x + w[1]  * x.y + w[2]  * x.z + w[3]  * x.w) * gv.x;
    y.y = (w[4]  * x.x + w[5]  * x.y + w[6]  * x.z + w[7]  * x.w) * gv.y;
    y.z = (w[8]  * x.x + w[9]  * x.y + w[10] * x.z + w[11] * x.w) * gv.z;
    y.w = (w[12] * x.x + w[13] * x.y + w[14] * x.z + w[15] * x.w) * gv.w;
    out[idx] = y;
}

// ---------------- host launch ----------------
extern "C" void kernel_launch(void* const* d_in, const int* in_sizes, int n_in,
                              void* d_out, int out_size, void* d_ws, size_t ws_size,
                              hipStream_t stream) {
    (void)in_sizes; (void)n_in; (void)out_size; (void)ws_size;
    const float* x1d = (const float*)d_in[0];
    const float* x2d = (const float*)d_in[1];
    const float* W1  = (const float*)d_in[2];
    const float* b1  = (const float*)d_in[3];
    const float* W2  = (const float*)d_in[4];
    const float* b2  = (const float*)d_in[5];
    const float* Wie = (const float*)d_in[6];
    const float* Wt  = (const float*)d_in[7];
    const float* Wg  = (const float*)d_in[8];
    const float* bg  = (const float*)d_in[9];

    float* x1d_out = (float*)d_out;                 // [64, 2048]
    float* x2d_out = x1d_out + NB * REPW;           // [64, 512, 512, 4]

    char* ws = (char*)d_ws;
    float*    norm = (float*)ws;    ws += NB * NCH * 4;
    _Float16* nhat = (_Float16*)ws; ws += NB * NCH * 2;
    _Float16* h1   = (_Float16*)ws; ws += NB * NCH * 2;
    float*    nmlp = (float*)ws;    ws += NB * NCH * 4;
    _Float16* gbuf = (_Float16*)ws; ws += NB * REPW * 2;
    _Float16* W1h  = (_Float16*)ws; ws += NCH * NCH * 2;
    _Float16* W2h  = (_Float16*)ws; ws += NCH * NCH * 2;
    _Float16* Wieh = (_Float16*)ws; ws += 4 * NCL * NCL * 2;
    float*    a2   = (float*)ws;    ws += NB * 4 * 4;
    float*    gate = (float*)ws;    ws += NB * 4 * 4;

    // a2 accumulated via atomics -> must be zeroed every call (graph-capture safe).
    hipMemsetAsync(a2, 0, NB * 4 * sizeof(float), stream);

    // fp16 weight copies
    cvt_k<<<(NCH * NCH + 255) / 256, 256, 0, stream>>>(W1, W1h, NCH * NCH);
    cvt_k<<<(NCH * NCH + 255) / 256, 256, 0, stream>>>(W2, W2h, NCH * NCH);
    cvt_k<<<(4 * NCL * NCL + 255) / 256, 256, 0, stream>>>(Wie, Wieh, 4 * NCL * NCL);

    // invariants + gauge
    repnorm_k<<<(NB * NCH) / 256, 256, 0, stream>>>(x1d, norm);
    bn_k<<<2, 256, 0, stream>>>(norm, nhat);
    gauge_k<<<(NB * REPW) / 256, 256, 0, stream>>>(x1d, norm, gbuf);

    // onebody MLP (WMMA): swish(n@W1^T+b1) @ W2^T + b2
    mlp_gemm_k<<<(NB / 16) * (NCH / 16), 32, 0, stream>>>(nhat, W1h, b1, h1, nullptr,
                                                          NCH, NCH, 1);
    mlp_gemm_k<<<(NB / 16) * (NCH / 16), 32, 0, stream>>>(h1, W2h, b2, nullptr, nmlp,
                                                          NCH, NCH, 0);

    // IELin (WMMA) + fused scalar_mul -> x1d_new in d_out
    ielin_k<<<512, 32, 0, stream>>>(gbuf, Wieh, nmlp, x1d_out);

    // twobody gated bilinear: two streaming passes over x2d (memory-bound floor)
    bilin1_k<<<NB * DFOLD, 256, 0, stream>>>((const float4*)x2d, (const float4*)x1d_out, a2);
    gate_k<<<1, 256, 0, stream>>>(a2, Wg, bg, gate);
    bilin2_k<<<(NB * DFOLD * DFOLD) / 256, 256, 0, stream>>>((const float4*)x2d, Wt,
                                                             (const float4*)gate,
                                                             (float4*)x2d_out);
}
